// Linear8bitLtOnline_9964324127140
// MI455X (gfx1250) — 8-bit matrix operand order repaired and checked by reading, compiled, not run on hardware
//
#include <hip/hip_runtime.h>
#include <hip/hip_bf16.h>

typedef int v8i __attribute__((ext_vector_type(8)));
typedef int vint4 __attribute__((vector_size(16)));

#define KO_MAX 32
#define BM 128
#define BN 128
#define BK 64
#define PITCH 80
#define TILEB (BM * PITCH)

#ifndef __has_builtin
#define __has_builtin(x) 0
#endif
#if __has_builtin(__builtin_amdgcn_global_load_async_to_lds_b128)
#define USE_ASYNC_LDS 1
#else
#define USE_ASYNC_LDS 0
#endif

#define ASRC(p) ((__attribute__((address_space(1))) vint4*)(const_cast<signed char*>(p)))
#define ADST(p) ((__attribute__((address_space(3))) vint4*)(p))

__global__ __launch_bounds__(256)
void init_kernel(unsigned* __restrict__ colmax_bits, int* __restrict__ cnt, int K) {
    int t = blockIdx.x * 256 + threadIdx.x;
    if (t < K) colmax_bits[t] = 0u;
    if (t == 0) *cnt = 0;
}

__global__ __launch_bounds__(256)
void colmax_kernel(const float* __restrict__ x, unsigned* __restrict__ colmax_bits,
                   int M, int K, int rows_per) {
    int k = blockIdx.x * 256 + threadIdx.x;
    if (k >= K) return;
    int r0 = blockIdx.y * rows_per;
    int r1 = r0 + rows_per; if (r1 > M) r1 = M;
    float mx = 0.f;
    for (int r = r0; r < r1; ++r)
        mx = fmaxf(mx, fabsf(x[(size_t)r * K + k]));
    atomicMax(colmax_bits + k, __float_as_uint(mx));
}

__global__ __launch_bounds__(256)
void detect_kernel(const unsigned* __restrict__ colmax_bits, const int* __restrict__ thr,
                   int* __restrict__ cnt, int* __restrict__ idx, int K) {
    int k = blockIdx.x * 256 + threadIdx.x;
    if (k >= K) return;
    float thrf = (float)(*thr);
    if (__uint_as_float(colmax_bits[k]) > thrf) {
        int p = atomicAdd(cnt, 1);
        if (p < KO_MAX) idx[p] = k;
    }
}

__global__ __launch_bounds__(256)
void gather_kernel(const float* __restrict__ x, const float* __restrict__ W,
                   const int* __restrict__ cnt, const int* __restrict__ idx,
                   float* __restrict__ xo, float* __restrict__ wo,
                   int M, int N, int K) {
    size_t t = (size_t)blockIdx.x * 256 + threadIdx.x;
    int row = (int)(t / KO_MAX);
    int j   = (int)(t % KO_MAX);
    if (row >= M + N) return;
    int c = *cnt; if (c > KO_MAX) c = KO_MAX;
    float v = 0.f;
    if (j < c) {
        int k = idx[j];
        v = (row < M) ? x[(size_t)row * K + k] : W[(size_t)(row - M) * K + k];
    }
    if (row < M) xo[(size_t)row * KO_MAX + j] = v;
    else         wo[(size_t)(row - M) * KO_MAX + j] = v;
}

__global__ __launch_bounds__(256)
void quant_x_kernel(const float* __restrict__ x, const unsigned* __restrict__ colmax_bits,
                    const int* __restrict__ thr, signed char* __restrict__ q,
                    float* __restrict__ srow, int K) {
    __shared__ float red[256];
    const int row = blockIdx.x, tid = threadIdx.x;
    const float* xr = x + (size_t)row * K;
    const float thrf = (float)(*thr);
    const int per = K >> 8;
    const int base = tid * per;
    float mx = 0.f;
    for (int i = 0; i < per; ++i) {
        int k = base + i;
        float v = xr[k];
        if (__uint_as_float(colmax_bits[k]) > thrf) v = 0.f;
        mx = fmaxf(mx, fabsf(v));
    }
    red[tid] = mx; __syncthreads();
    for (int s = 128; s > 0; s >>= 1) {
        if (tid < s) red[tid] = fmaxf(red[tid], red[tid + s]);
        __syncthreads();
    }
    float scale = fmaxf(red[0] * (1.f / 127.f), 1e-8f);
    float inv = 1.f / scale;
    if (tid == 0) srow[row] = scale;
    for (int i = 0; i < per; i += 4) {
        unsigned pack = 0;
        for (int j = 0; j < 4; ++j) {
            int k = base + i + j;
            float v = xr[k];
            if (__uint_as_float(colmax_bits[k]) > thrf) v = 0.f;
            float qf = fminf(fmaxf(rintf(v * inv), -127.f), 127.f);
            pack |= ((unsigned)((int)qf & 0xFF)) << (8 * j);
        }
        *(unsigned*)(q + (size_t)row * K + base + i) = pack;
    }
}

__global__ __launch_bounds__(256)
void quant_w_kernel(const float* __restrict__ W, signed char* __restrict__ q,
                    float* __restrict__ srow, int K) {
    __shared__ float red[256];
    const int row = blockIdx.x, tid = threadIdx.x;
    const float* wr = W + (size_t)row * K;
    const int per = K >> 8;
    const int base = tid * per;
    float mx = 0.f;
    for (int i = 0; i < per; ++i) mx = fmaxf(mx, fabsf(wr[base + i]));
    red[tid] = mx; __syncthreads();
    for (int s = 128; s > 0; s >>= 1) {
        if (tid < s) red[tid] = fmaxf(red[tid], red[tid + s]);
        __syncthreads();
    }
    float scale = fmaxf(red[0] * (1.f / 127.f), 1e-8f);
    float inv = 1.f / scale;
    if (tid == 0) srow[row] = scale;
    for (int i = 0; i < per; i += 4) {
        unsigned pack = 0;
        for (int j = 0; j < 4; ++j) {
            float qf = fminf(fmaxf(rintf(wr[base + i + j] * inv), -127.f), 127.f);
            pack |= ((unsigned)((int)qf & 0xFF)) << (8 * j);
        }
        *(unsigned*)(q + (size_t)row * K + base + i) = pack;
    }
}

__global__ __launch_bounds__(256)
void gemm_kernel(const signed char* __restrict__ qx, const signed char* __restrict__ qw,
                 const float* __restrict__ sx, const float* __restrict__ sw,
                 const float* __restrict__ xo, const float* __restrict__ wo,
                 const float* __restrict__ bias, float* __restrict__ out,
                 int M, int N, int K) {
    __shared__ signed char As[2 * TILEB];
    __shared__ signed char Bs[2 * TILEB];
    __shared__ float xos[BM * KO_MAX];

    const int tid  = threadIdx.x;
    const int lane = tid & 31;
    const int wv   = tid >> 5;
    const int half = lane >> 4;
    const int l16  = lane & 15;
    const int mbase = blockIdx.y * BM;
    const int nbase = blockIdx.x * BN;

    {
        const float4* gx = (const float4*)(xo + (size_t)mbase * KO_MAX);
        float4* lx = (float4*)xos;
        for (int i = tid; i < (BM * KO_MAX) / 4; i += 256) lx[i] = gx[i];
    }

    v8i acc[4][2];
#pragma unroll
    for (int mt = 0; mt < 4; ++mt)
#pragma unroll
        for (int nt = 0; nt < 2; ++nt)
            acc[mt][nt] = (v8i)0;

    const int lr = tid >> 1;
    const int lc = (tid & 1) * 32;
    const int rowA = (wv >> 2) * 64;
    const int colB = (wv & 3) * 32;

    const signed char* gA = qx + (size_t)(mbase + lr) * K + lc;
    const signed char* gB = qw + (size_t)(nbase + lr) * K + lc;
    const int ldsOff = lr * PITCH + lc;

#if USE_ASYNC_LDS
    {
        signed char* lA = As + ldsOff;
        signed char* lB = Bs + ldsOff;
        __builtin_amdgcn_global_load_async_to_lds_b128(ASRC(gA),      ADST(lA),      0, 0);
        __builtin_amdgcn_global_load_async_to_lds_b128(ASRC(gA + 16), ADST(lA + 16), 0, 0);
        __builtin_amdgcn_global_load_async_to_lds_b128(ASRC(gB),      ADST(lB),      0, 0);
        __builtin_amdgcn_global_load_async_to_lds_b128(ASRC(gB + 16), ADST(lB + 16), 0, 0);
    }
#else
    int4 ra0 = *(const int4*)(gA);
    int4 ra1 = *(const int4*)(gA + 16);
    int4 rb0 = *(const int4*)(gB);
    int4 rb1 = *(const int4*)(gB + 16);
#endif

    int p = 0;
    for (int k0 = 0; k0 < K; k0 += BK, p ^= 1) {
        const int has_next = (k0 + BK < K);

#if USE_ASYNC_LDS
        if (has_next) {
            signed char* lA = As + (p ^ 1) * TILEB + ldsOff;
            signed char* lB = Bs + (p ^ 1) * TILEB + ldsOff;
            const signed char* nA = gA + k0 + BK;
            const signed char* nB = gB + k0 + BK;
            __builtin_amdgcn_global_load_async_to_lds_b128(ASRC(nA),      ADST(lA),      0, 0);
            __builtin_amdgcn_global_load_async_to_lds_b128(ASRC(nA + 16), ADST(lA + 16), 0, 0);
            __builtin_amdgcn_global_load_async_to_lds_b128(ASRC(nB),      ADST(lB),      0, 0);
            __builtin_amdgcn_global_load_async_to_lds_b128(ASRC(nB + 16), ADST(lB + 16), 0, 0);
            if (k0 + 2 * BK < K) {
                __builtin_prefetch(gA + k0 + 2 * BK, 0, 3);
                __builtin_prefetch(gB + k0 + 2 * BK, 0, 3);
            }
            asm volatile("s_wait_asynccnt 0x4" ::: "memory");
        } else {
            asm volatile("s_wait_asynccnt 0x0" ::: "memory");
        }
        __syncthreads();
#else
        __syncthreads();
        *(int4*)(As + p * TILEB + ldsOff)      = ra0;
        *(int4*)(As + p * TILEB + ldsOff + 16) = ra1;
        *(int4*)(Bs + p * TILEB + ldsOff)      = rb0;
        *(int4*)(Bs + p * TILEB + ldsOff + 16) = rb1;
        __syncthreads();
        if (has_next) {
            ra0 = *(const int4*)(gA + k0 + BK);
            ra1 = *(const int4*)(gA + k0 + BK + 16);
            rb0 = *(const int4*)(gB + k0 + BK);
            rb1 = *(const int4*)(gB + k0 + BK + 16);
            if (k0 + 2 * BK < K) {
                __builtin_prefetch(gA + k0 + 2 * BK, 0, 3);
                __builtin_prefetch(gB + k0 + 2 * BK, 0, 3);
            }
        }
#endif

        const signed char* Ab = As + p * TILEB;
        const signed char* Bb = Bs + p * TILEB;

        v8i af[4], bf[2];
#pragma unroll
        for (int mt = 0; mt < 4; ++mt) {
            const signed char* pp = Ab + (rowA + mt * 16 + l16) * PITCH + half * 16;
            int4 q0 = *(const int4*)(pp +  0);
            int4 q1 = *(const int4*)(pp + 32);
            af[mt][0] = q0.x; af[mt][1] = q0.y; af[mt][2] = q0.z; af[mt][3] = q0.w;
            af[mt][4] = q1.x; af[mt][5] = q1.y; af[mt][6] = q1.z; af[mt][7] = q1.w;
        }
#pragma unroll
        for (int nt = 0; nt < 2; ++nt) {
            const signed char* pp = Bb + (colB + nt * 16 + l16) * PITCH + half * 16;
            int4 q0 = *(const int4*)(pp +  0);
            int4 q1 = *(const int4*)(pp + 32);
            bf[nt][0] = q0.x; bf[nt][1] = q0.y; bf[nt][2] = q0.z; bf[nt][3] = q0.w;
            bf[nt][4] = q1.x; bf[nt][5] = q1.y; bf[nt][6] = q1.z; bf[nt][7] = q1.w;
        }
#pragma unroll
        for (int mt = 0; mt < 4; ++mt)
#pragma unroll
            for (int nt = 0; nt < 2; ++nt)
                acc[mt][nt] = __builtin_amdgcn_wmma_i32_16x16x64_iu8(
                    true, af[mt], true, bf[nt], acc[mt][nt], false, false);

        __syncthreads();
    }

    float wrow[2][KO_MAX];
    float swv[2], bv[2];
    int nn[2];
#pragma unroll
    for (int nt = 0; nt < 2; ++nt) {
        int nl = colB + nt * 16 + l16;
        nn[nt] = nbase + nl;
        swv[nt] = sw[nn[nt]];
        bv[nt]  = bias[nn[nt]];
        const float4* wp = (const float4*)(wo + (size_t)nn[nt] * KO_MAX);
#pragma unroll
        for (int j = 0; j < KO_MAX / 4; ++j) {
            float4 f = wp[j];
            wrow[nt][4*j+0] = f.x; wrow[nt][4*j+1] = f.y;
            wrow[nt][4*j+2] = f.z; wrow[nt][4*j+3] = f.w;
        }
    }
#pragma unroll
    for (int mt = 0; mt < 4; ++mt) {
#pragma unroll
        for (int v = 0; v < 8; ++v) {
            int ml = rowA + mt * 16 + half * 8 + v;
            int m  = mbase + ml;
            float sxv = sx[m];
            float xrow[KO_MAX];
            const float4* xp = (const float4*)(xos + ml * KO_MAX);
#pragma unroll
            for (int j = 0; j < KO_MAX / 4; ++j) {
                float4 f = xp[j];
                xrow[4*j+0] = f.x; xrow[4*j+1] = f.y;
                xrow[4*j+2] = f.z; xrow[4*j+3] = f.w;
            }
#pragma unroll
            for (int nt = 0; nt < 2; ++nt) {
                float dot = 0.f;
#pragma unroll
                for (int j = 0; j < KO_MAX; ++j) dot = fmaf(xrow[j], wrow[nt][j], dot);
                out[(size_t)m * N + nn[nt]] =
                    (float)acc[mt][nt][v] * sxv * swv[nt] + dot + bv[nt];
            }
        }
    }
}

extern "C" void kernel_launch(void* const* d_in, const int* in_sizes, int n_in,
                              void* d_out, int out_size, void* d_ws, size_t ws_size,
                              hipStream_t stream) {
    const float* x    = (const float*)d_in[0];
    const float* W    = (const float*)d_in[1];
    const float* bias = (const float*)d_in[2];
    const int*   thr  = (const int*)d_in[3];

    const int N = in_sizes[2];
    const int K = in_sizes[1] / N;
    const int M = in_sizes[0] / K;
    (void)n_in; (void)out_size; (void)ws_size;

    char* ws = (char*)d_ws;
    size_t cur = 0;
    auto take = [&](size_t bytes) -> char* {
        cur = (cur + 255) & ~(size_t)255;
        char* p = ws + cur;
        cur += bytes;
        return p;
    };
    signed char* qx  = (signed char*)take((size_t)M * K);
    signed char* qw  = (signed char*)take((size_t)N * K);
    float* sx        = (float*)take((size_t)M * 4);
    float* sw        = (float*)take((size_t)N * 4);
    unsigned* colmax = (unsigned*)take((size_t)K * 4);
    int* cnt         = (int*)take(16);
    int* idx         = (int*)take(KO_MAX * 4);
    float* xo        = (float*)take((size_t)M * KO_MAX * 4);
    float* wo        = (float*)take((size_t)N * KO_MAX * 4);

    const int kb = (K + 255) / 256;
    init_kernel<<<kb, 256, 0, stream>>>(colmax, cnt, K);

    const int rows_per = 256;
    dim3 cg(kb, (M + rows_per - 1) / rows_per);
    colmax_kernel<<<cg, 256, 0, stream>>>(x, colmax, M, K, rows_per);

    detect_kernel<<<kb, 256, 0, stream>>>(colmax, thr, cnt, idx, K);

    size_t gtot = (size_t)(M + N) * KO_MAX;
    gather_kernel<<<(int)((gtot + 255) / 256), 256, 0, stream>>>(x, W, cnt, idx, xo, wo, M, N, K);

    quant_x_kernel<<<M, 256, 0, stream>>>(x, colmax, thr, qx, sx, K);
    quant_w_kernel<<<N, 256, 0, stream>>>(W, qw, sw, K);

    dim3 gg(N / BN, M / BM);
    gemm_kernel<<<gg, 256, 0, stream>>>(qx, qw, sx, sw, xo, wo, bias, (float*)d_out, M, N, K);
}
